// LightningIndexer_19542101197223
// MI455X (gfx1250) — compile-verified, matrix operand order checked by reading, not run on hardware
//
#include <hip/hip_runtime.h>

typedef __attribute__((ext_vector_type(16))) __bf16 v16bf;
typedef __attribute__((ext_vector_type(8)))  float  v8f;

#define B_   2
#define T_   4096
#define D_   2048
#define H_   4
#define DH_  32
#define NROWS (B_ * T_)        // 8192 flat (b,t) rows
#define NCOLS 164              // 128 (q) + 4 (w) + 32 (k)
#define NCOLS_PAD 176          // 11 * 16
#define NT_TILES 11

union BF16Vec {
    v16bf v;
    __bf16 b[16];
    uint4  q[2];
};

// Native f32 -> bf16 (RNE) via hardware convert.
__device__ __forceinline__ void cvt8_bf(const float* __restrict__ p, __bf16* dst) {
    float4 f0 = *(const float4*)p;
    float4 f1 = *(const float4*)(p + 4);
    dst[0] = (__bf16)f0.x; dst[1] = (__bf16)f0.y;
    dst[2] = (__bf16)f0.z; dst[3] = (__bf16)f0.w;
    dst[4] = (__bf16)f1.x; dst[5] = (__bf16)f1.y;
    dst[6] = (__bf16)f1.z; dst[7] = (__bf16)f1.w;
}

// ---------------------------------------------------------------------------
// Kernel 0: one-shot weight concat + f32->bf16 convert.
//   Wcat[176][2048] bf16; rows [0,128)=Wq, [128,132)=Ww, [132,164)=Wk,
//   rows [164,176) = zero pad. 8 elements per thread.
// ---------------------------------------------------------------------------
__global__ __launch_bounds__(256) void li_wcat_kernel(
    const float* __restrict__ Wq, const float* __restrict__ Ww,
    const float* __restrict__ Wk, __bf16* __restrict__ wcat)
{
    const int tid = blockIdx.x * 256 + threadIdx.x;   // 176*256 threads
    const int row = tid >> 8;                          // [0,176)
    const int c0  = (tid & 255) * 8;                   // [0,2048) step 8

    __bf16* dst = wcat + (size_t)row * D_ + c0;
    if (row < 128) {
        cvt8_bf(Wq + (size_t)row * D_ + c0, dst);
    } else if (row < 132) {
        cvt8_bf(Ww + (size_t)(row - 128) * D_ + c0, dst);
    } else if (row < 164) {
        cvt8_bf(Wk + (size_t)(row - 132) * D_ + c0, dst);
    } else {
#pragma unroll
        for (int i = 0; i < 8; ++i) dst[i] = (__bf16)0.0f;
    }
}

// ---------------------------------------------------------------------------
// Kernel 1: fused projection GEMM  Y(8192 x 164) = X(8192 x 2048) @ Wcat^T
// One wave per 16x16 output tile; K = 2048 -> 64 bf16 WMMAs.
// A (hidden) converted f32->bf16 in registers; B loaded as bf16 directly.
// ---------------------------------------------------------------------------
__global__ __launch_bounds__(256) void li_proj_kernel(
    const float* __restrict__ hidden, const __bf16* __restrict__ wcat,
    __bf16* __restrict__ qb, float* __restrict__ wf, __bf16* __restrict__ kb)
{
    const int lane = threadIdx.x & 31;
    const int wave = blockIdx.x * 8 + (threadIdx.x >> 5);   // 5632 waves exactly
    const int rt   = wave / NT_TILES;                        // row tile [0,512)
    const int nt   = wave % NT_TILES;                        // col tile [0,11)

    const int g  = lane >> 4;      // half-wave group (K-chunk select)
    const int mr = lane & 15;      // A row / B column within tile

    const float*  aptr = hidden + (size_t)(rt * 16 + mr) * D_;
    const int     ncol = nt * 16 + mr;                       // [0,176), pad-safe
    const __bf16* bptr = wcat + (size_t)ncol * D_;

    v8f acc = {};
    for (int kk = 0; kk < D_; kk += 32) {
        BF16Vec a, bm;
        cvt8_bf(aptr + kk + 8 * g,      &a.b[0]);            // K = kk+8g..+7
        cvt8_bf(aptr + kk + 16 + 8 * g, &a.b[8]);            // K = kk+16+8g..+7
        bm.q[0] = *(const uint4*)(bptr + kk + 8 * g);
        bm.q[1] = *(const uint4*)(bptr + kk + 16 + 8 * g);
        acc = __builtin_amdgcn_wmma_f32_16x16x32_bf16(
                  false, a.v, false, bm.v, (short)0, acc, false, false);
    }

    // D element r of this lane is output row rt*16 + r + 8g, column ncol.
#pragma unroll
    for (int r = 0; r < 8; ++r) {
        const int row = rt * 16 + r + 8 * g;
        const float val = acc[r];
        if (ncol < 128) {
            qb[(size_t)row * (H_ * DH_) + ncol] = (__bf16)val;
        } else if (ncol < 132) {
            wf[(size_t)row * H_ + (ncol - 128)] = val;
        } else if (ncol < 164) {
            kb[(size_t)row * DH_ + (ncol - 132)] = (__bf16)val;
        }
    }
}

// ---------------------------------------------------------------------------
// Kernel 2: scores.  Per wave: 16 t-rows x 64 s-cols.
//   acc_h(16x16) = Q_h(16x32) x K^T(32x16)   -- one bf16 WMMA per head
//   out = sum_h w[t,h] * relu(acc_h)         -- f32 VALU epilogue
// ---------------------------------------------------------------------------
__global__ __launch_bounds__(256) void li_score_kernel(
    const __bf16* __restrict__ qb, const float* __restrict__ wf,
    const __bf16* __restrict__ kb, float* __restrict__ out)
{
    const int lane = threadIdx.x & 31;
    const int wave = blockIdx.x * 8 + (threadIdx.x >> 5);  // 32768 waves exactly
    const int sg = wave & 63;            // 64-col group   [0,64)
    const int tt = (wave >> 6) & 255;    // 16-row tile    [0,256)
    const int b  = wave >> 14;           // batch          [0,2)

    const int g  = lane >> 4;
    const int mr = lane & 15;
    const int t0 = tt * 16;
    const int s0 = sg * 64;

    // A: four 16x32 bf16 tiles (one per head), row t0+mr, reused 4 subtiles.
    BF16Vec a[H_];
    {
        const __bf16* qrow = qb + (size_t)(b * T_ + t0 + mr) * (H_ * DH_);
#pragma unroll
        for (int h = 0; h < H_; ++h) {
            const __bf16* p = qrow + h * DH_;
            a[h].q[0] = *(const uint4*)(p + 8 * g);
            a[h].q[1] = *(const uint4*)(p + 16 + 8 * g);
        }
    }

    // Per-row head weights: wv[h][r] for output row t0 + r + 8g.
    float wv[H_][8];
#pragma unroll
    for (int r = 0; r < 8; ++r) {
        const float* wp = wf + (size_t)(b * T_ + t0 + r + 8 * g) * H_;
#pragma unroll
        for (int h = 0; h < H_; ++h) wv[h][r] = wp[h];
    }

    for (int ss = 0; ss < 4; ++ss) {
        const int sb = s0 + ss * 16;

        // B: K^T tile. Column n = sb+mr of K^T is row sb+mr of k (row-major [*,32]).
        BF16Vec bm;
        const __bf16* krow = kb + (size_t)(b * T_ + sb + mr) * DH_;
        bm.q[0] = *(const uint4*)(krow + 8 * g);
        bm.q[1] = *(const uint4*)(krow + 16 + 8 * g);

        v8f acc[H_];
#pragma unroll
        for (int h = 0; h < H_; ++h) {
            v8f z = {};
            acc[h] = __builtin_amdgcn_wmma_f32_16x16x32_bf16(
                         false, a[h].v, false, bm.v, (short)0, z, false, false);
        }

        // ReLU + head-weighted combine, coalesced f32 stores.
        float* op = out + (size_t)b * T_ * T_ + (size_t)(t0 + 8 * g) * T_ + sb + mr;
#pragma unroll
        for (int r = 0; r < 8; ++r) {
            float v = 0.0f;
#pragma unroll
            for (int h = 0; h < H_; ++h)
                v = fmaf(wv[h][r], fmaxf(acc[h][r], 0.0f), v);
            op[(size_t)r * T_] = v;
        }
    }
}

extern "C" void kernel_launch(void* const* d_in, const int* in_sizes, int n_in,
                              void* d_out, int out_size, void* d_ws, size_t ws_size,
                              hipStream_t stream) {
    (void)in_sizes; (void)n_in; (void)out_size; (void)ws_size;
    const float* hidden = (const float*)d_in[0];
    const float* Wq     = (const float*)d_in[1];
    const float* Ww     = (const float*)d_in[2];
    const float* Wk     = (const float*)d_in[3];
    float* out = (float*)d_out;

    // Workspace layout:
    //   qb   @ 0       : 8192*128 bf16 = 2 MB
    //   wf   @ 2 MB    : 8192*4   f32  = 128 KB
    //   kb   @ 2.125 MB: 8192*32  bf16 = 512 KB
    //   wcat @ 2.625 MB: 176*2048 bf16 = 704 KB
    char* ws = (char*)d_ws;
    __bf16* qb   = (__bf16*)ws;
    float*  wf   = (float*)(ws + (size_t)2 * 1024 * 1024);
    __bf16* kb   = (__bf16*)(ws + (size_t)2 * 1024 * 1024 + 128 * 1024);
    __bf16* wcat = (__bf16*)(ws + (size_t)2 * 1024 * 1024 + 640 * 1024);

    // 176 rows * 256 chunks-of-8 = 45056 threads -> 176 blocks.
    li_wcat_kernel<<<176, 256, 0, stream>>>(Wq, Ww, Wk, wcat);
    // 512 row-tiles * 11 col-tiles = 5632 waves -> 704 blocks of 8 waves.
    li_proj_kernel<<<704, 256, 0, stream>>>(hidden, wcat, qb, wf, kb);
    // 2 * 256 * 64 = 32768 waves -> 4096 blocks of 8 waves.
    li_score_kernel<<<4096, 256, 0, stream>>>(qb, wf, kb, out);
}
